// NetVLADLayer_26706106646880
// MI455X (gfx1250) — matrix operand order repaired and checked by reading, not compiled, not run on hardware
//
#include <hip/hip_runtime.h>

#define BATCH   32
#define DDIM    512
#define NDIM    2048
#define KDIM    64
#define NTILE   64
#define NCHUNK  (NDIM / NTILE)   // 32
#define THREADS 512              // 16 waves (wave32)
#define EPSN    1e-12f

// padded LDS strides (elements) to kill bank conflicts
#define WSTR   520   // sW / sXt rows: 1040B -> 4-bank shift per row
#define XSTR   72    // sX / Sbf rows: 36 dwords, gcd(36,64)=4 -> distinct groups
#define SSTR   65    // S rows: conflict-free column access

typedef __attribute__((ext_vector_type(8)))  __bf16 v8bf;
typedef __attribute__((ext_vector_type(16))) __bf16 v16bf;
typedef __attribute__((ext_vector_type(8)))  float  v8f;

// ---- LDS layout (bytes) ----
#define OFF_W    0                        // [64][520] bf16  (66560)
#define OFF_XT   66560                    // [64][520] bf16  (66560)  sXt[n][d]
#define OFF_X    133120                   // [512][72] bf16  (73728)  sX[d][n]
#define OFF_S    206848                   // [64][65]  f32   (16640)  S[k][n]
#define OFF_SBF  223488                   // [64][72]  bf16  (9216)
#define OFF_SSUM 232704                   // [64] f32
#define OFF_PART 232960                   // [512] f32
#define OFF_INVK 235008                   // [64] f32
#define OFF_GS   235264                   // f32
#define SMEM_BYTES 235520
// descBuf [512][64] f32 (131072) overlays sW+sXt after the GEMM loop

__device__ __forceinline__ unsigned short f2bfu(float f) {
  union { float f; unsigned u; } v; v.f = f;
  unsigned r = v.u + 0x7FFFu + ((v.u >> 16) & 1u);   // round-to-nearest-even
  return (unsigned short)(r >> 16);
}
__device__ __forceinline__ __bf16 u2bf(unsigned short h) {
  return __builtin_bit_cast(__bf16, h);
}
__device__ __forceinline__ v16bf cat16(v8bf lo, v8bf hi) {
  return __builtin_shufflevector(lo, hi, 0,1,2,3,4,5,6,7,8,9,10,11,12,13,14,15);
}

__global__ __launch_bounds__(THREADS, 1)
void netvlad_fused_kernel(const float* __restrict__ x,
                          const float* __restrict__ W,
                          const float* __restrict__ centers,
                          float* __restrict__ out) {
  extern __shared__ char smem[];
  __bf16* sW      = (__bf16*)(smem + OFF_W);
  __bf16* sXt     = (__bf16*)(smem + OFF_XT);
  __bf16* sX      = (__bf16*)(smem + OFF_X);
  float*  S       = (float*) (smem + OFF_S);
  __bf16* Sbf     = (__bf16*)(smem + OFF_SBF);
  float*  ssum    = (float*) (smem + OFF_SSUM);
  float*  partial = (float*) (smem + OFF_PART);
  float*  invk    = (float*) (smem + OFF_INVK);
  float*  gscale  = (float*) (smem + OFF_GS);
  float*  descBuf = (float*) (smem);

  const int t    = threadIdx.x;
  const int lane = t & 31;
  const int wave = t >> 5;
  const int half = lane >> 4;     // 0/1: selects K sub-range per WMMA layouts
  const int ln16 = lane & 15;
  const int b    = blockIdx.x;
  const float* xb = x + (size_t)b * DDIM * NDIM;

  // ---- Stage W as bf16 in LDS (once) ----
  for (int i = 0; i < 16; ++i) {
    int e = (t + THREADS * i) * 4;                 // 32768 floats total
    float4 w4 = *(const float4*)(W + e);
    unsigned lo = (unsigned)f2bfu(w4.x) | ((unsigned)f2bfu(w4.y) << 16);
    unsigned hi = (unsigned)f2bfu(w4.z) | ((unsigned)f2bfu(w4.w) << 16);
    *(uint2*)(sW + (e >> 9) * WSTR + (e & 511)) = make_uint2(lo, hi);
  }
  if (t < KDIM) ssum[t] = 0.f;

  // persistent agg accumulators: wave owns 8 (d,k) 16x16 tiles
  v8f acc[8];
  #pragma unroll
  for (int i = 0; i < 8; ++i) acc[i] = v8f{};

  const int kt   = wave & 3;   // step C: k-tile
  const int dgrp = wave >> 2;  // step C: d-tile group (8 tiles each)

  // ---- software pipeline: stage chunk 0 into registers ----
  float4 reg[16];
  #pragma unroll
  for (int i = 0; i < 16; ++i) {
    int e = (t + THREADS * i) * 4;
    reg[i] = *(const float4*)(xb + (size_t)(e >> 6) * NDIM + (e & 63));
  }
  __syncthreads();

  for (int nc = 0; nc < NCHUNK; ++nc) {
    const int n0 = nc * NTILE;

    // ---- drain register stage: sX[d][n] and sXt[n][d], bf16 ----
    #pragma unroll
    for (int i = 0; i < 16; ++i) {
      int e = (t + THREADS * i) * 4;               // 512*64 floats
      int d = e >> 6;
      int j = e & 63;
      float4 v = reg[i];
      unsigned short h0 = f2bfu(v.x), h1 = f2bfu(v.y),
                     h2 = f2bfu(v.z), h3 = f2bfu(v.w);
      unsigned lo = (unsigned)h0 | ((unsigned)h1 << 16);
      unsigned hi = (unsigned)h2 | ((unsigned)h3 << 16);
      *(uint2*)(sX + d * XSTR + j) = make_uint2(lo, hi);
      sXt[(j + 0) * WSTR + d] = u2bf(h0);
      sXt[(j + 1) * WSTR + d] = u2bf(h1);
      sXt[(j + 2) * WSTR + d] = u2bf(h2);
      sXt[(j + 3) * WSTR + d] = u2bf(h3);
    }
    __syncthreads();

    // ---- issue next chunk's global loads; in flight during the GEMMs ----
    if (nc + 1 < NCHUNK) {
      #pragma unroll
      for (int i = 0; i < 16; ++i) {
        int e = (t + THREADS * i) * 4;
        reg[i] = *(const float4*)(xb + (size_t)(e >> 6) * NDIM +
                                  (n0 + NTILE) + (e & 63));
      }
      if (nc + 2 < NCHUNK)   // warm L2 one chunk further (global_prefetch_b8)
        __builtin_prefetch(xb + (size_t)t * NDIM + n0 + 2 * NTILE, 0, 1);
    }

    // ---- Step A: S(64x64) = W(64x512) * Xc(512x64); one 16x16 tile per wave
    {
      const int mt = wave & 3, nt = wave >> 2;
      v8f c = v8f{};
      const __bf16* wrow = sW  + (size_t)(mt * 16 + ln16) * WSTR; // A row = K idx
      const __bf16* xrow = sXt + (size_t)(nt * 16 + ln16) * WSTR; // B col = n idx
      #pragma unroll
      for (int kd = 0; kd < DDIM; kd += 32) {
        v8bf alo = *(const v8bf*)(wrow + kd + 8 * half);
        v8bf ahi = *(const v8bf*)(wrow + kd + 16 + 8 * half);
        v16bf a  = cat16(alo, ahi);
        v16bf bb = cat16(*(const v8bf*)(xrow + kd + 8 * half),
                         *(const v8bf*)(xrow + kd + 16 + 8 * half));
        c = __builtin_amdgcn_wmma_f32_16x16x32_bf16(false, a, false, bb,
                                                    (short)0, c, false, false);
      }
      #pragma unroll
      for (int r = 0; r < 8; ++r)
        S[(mt * 16 + r + 8 * half) * SSTR + nt * 16 + ln16] = c[r];
    }
    __syncthreads();

    // ---- softmax over K per column n (threads 0..63), write bf16 copy ----
    if (t < NTILE) {
      const int n = t;
      float mx = -1e30f;
      for (int k = 0; k < KDIM; ++k) mx = fmaxf(mx, S[k * SSTR + n]);
      float sum = 0.f;
      for (int k = 0; k < KDIM; ++k) {
        float e = __expf(S[k * SSTR + n] - mx);
        S[k * SSTR + n] = e;
        sum += e;
      }
      float inv = 1.f / sum;
      for (int k = 0; k < KDIM; ++k) {
        float v = S[k * SSTR + n] * inv;
        S[k * SSTR + n]   = v;
        Sbf[k * XSTR + n] = u2bf(f2bfu(v));
      }
    }
    __syncthreads();

    // s_sum[k] += row-sum over this chunk (threads 0..63, own row each)
    if (t < KDIM) {
      float rs = 0.f;
      for (int n = 0; n < NTILE; ++n) rs += S[t * SSTR + n];
      ssum[t] += rs;
    }

    // ---- Step C: agg(512x64) += Xc(512x64) * S^T(64x64), reg-resident C ----
    #pragma unroll
    for (int i = 0; i < 8; ++i) {
      const int dt = dgrp * 8 + i;
      const __bf16* xrow = sX  + (size_t)(dt * 16 + ln16) * XSTR; // A row = d
      const __bf16* srow = Sbf + (size_t)(kt * 16 + ln16) * XSTR; // B col = k
      #pragma unroll
      for (int ks = 0; ks < 2; ++ks) {                            // n-reduction
        v8bf alo = *(const v8bf*)(xrow + ks * 32 + 8 * half);
        v8bf ahi = *(const v8bf*)(xrow + ks * 32 + 16 + 8 * half);
        v16bf a  = cat16(alo, ahi);
        v16bf bb = cat16(*(const v8bf*)(srow + ks * 32 + 8 * half),
                         *(const v8bf*)(srow + ks * 32 + 16 + 8 * half));
        acc[i] = __builtin_amdgcn_wmma_f32_16x16x32_bf16(false, a, false, bb,
                                                         (short)0, acc[i],
                                                         false, false);
      }
    }
    __syncthreads();
  }

  // ---- Epilogue: desc = agg - centers * s_sum  -> descBuf (overlays sW/sXt)
  #pragma unroll
  for (int i = 0; i < 8; ++i) {
    const int dt = dgrp * 8 + i;
    const int k  = kt * 16 + ln16;
    #pragma unroll
    for (int r = 0; r < 8; ++r) {
      const int d = dt * 16 + r + 8 * half;
      float v = acc[i][r] - centers[d * KDIM + k] * ssum[k];
      descBuf[d * 64 + k] = v;
    }
  }
  __syncthreads();

  // per-k sum of squares over D (8 partials per k)
  {
    const int k = t & 63, part = t >> 6;
    float s = 0.f;
    for (int d = part * 64; d < part * 64 + 64; ++d) {
      float v = descBuf[d * 64 + k];
      s += v * v;
    }
    partial[part * 64 + k] = s;
  }
  __syncthreads();
  if (t < KDIM) {
    float ss = 0.f;
    for (int p = 0; p < 8; ++p) ss += partial[p * 64 + t];
    float iv = 1.f / fmaxf(sqrtf(ss), EPSN);
    invk[t] = iv;
    partial[t] = ss * iv * iv;   // contribution to global sumsq
  }
  __syncthreads();
  if (t == 0) {
    float g = 0.f;
    for (int k = 0; k < KDIM; ++k) g += partial[k];
    gscale[0] = 1.f / fmaxf(sqrtf(g), EPSN);
  }
  __syncthreads();

  // final store: out[b, d*K + k], coalesced
  {
    const float gs = gscale[0];
    const float ik = invk[t & 63];
    float* ob = out + (size_t)b * (DDIM * KDIM);
    for (int i = 0; i < 64; ++i) {
      int flat = i * THREADS + t;
      ob[flat] = descBuf[flat] * ik * gs;
    }
  }
}

extern "C" void kernel_launch(void* const* d_in, const int* in_sizes, int n_in,
                              void* d_out, int out_size, void* d_ws, size_t ws_size,
                              hipStream_t stream) {
  (void)in_sizes; (void)n_in; (void)out_size; (void)d_ws; (void)ws_size;
  const float* x       = (const float*)d_in[0];
  const float* W       = (const float*)d_in[1];
  const float* centers = (const float*)d_in[2];
  float* out = (float*)d_out;
  netvlad_fused_kernel<<<dim3(BATCH), dim3(THREADS), SMEM_BYTES, stream>>>(
      x, W, centers, out);
}
